// RefNetLoRAProcessor_58875411694123
// MI455X (gfx1250) — compile-verified, matrix operand order checked by reading, not run on hardware
//
#include <hip/hip_runtime.h>
#include <hip/hip_bf16.h>

typedef __bf16 bf16;
typedef __attribute__((ext_vector_type(16))) __bf16 bf16x16;
typedef __attribute__((ext_vector_type(8)))  __bf16 bf16x8;
typedef __attribute__((ext_vector_type(8)))  float  f32x8;
typedef __attribute__((ext_vector_type(4)))  int    v4i_t;

#define S_TOT 2048
#define TXT   226
#define VID   1822
#define DIM   3072
#define NQKV  9216
#define RANK  128
#define NHEAD 48
#define HDIM  64

#define AS1 __attribute__((address_space(1)))
#define AS3 __attribute__((address_space(3)))

#if __has_builtin(__builtin_amdgcn_global_load_async_to_lds_b128)
#define ASYNC_LDS 1
#else
#define ASYNC_LDS 0
#endif

static __device__ __forceinline__ bf16x16 cat8(bf16x8 lo, bf16x8 hi) {
  return __builtin_shufflevector(lo, hi, 0,1,2,3,4,5,6,7,8,9,10,11,12,13,14,15);
}
static __device__ __forceinline__ f32x8 wmma_bf16(bf16x16 a, bf16x16 b, f32x8 c) {
  // 8 args: (neg_a, A, neg_b, B, c_mod, C, reuse_a, reuse_b)
  return __builtin_amdgcn_wmma_f32_16x16x32_bf16(false, a, false, b, (short)0, c, false, false);
}
static __device__ __forceinline__ float redmax16(float v) {
  v = fmaxf(v, __shfl_xor(v, 1, 32));
  v = fmaxf(v, __shfl_xor(v, 2, 32));
  v = fmaxf(v, __shfl_xor(v, 4, 32));
  v = fmaxf(v, __shfl_xor(v, 8, 32));
  return v;
}
static __device__ __forceinline__ float redsum16(float v) {
  v += __shfl_xor(v, 1, 32);
  v += __shfl_xor(v, 2, 32);
  v += __shfl_xor(v, 4, 32);
  v += __shfl_xor(v, 8, 32);
  return v;
}
#if ASYNC_LDS
static __device__ __forceinline__ void async_ld16(const bf16* g, bf16* l) {
  // GLOBAL_LOAD_ASYNC_TO_LDS_B128: DMA 16B/lane global -> LDS, tracked by ASYNCcnt
  __builtin_amdgcn_global_load_async_to_lds_b128((AS1 v4i_t*)g, (AS3 v4i_t*)l, 0, 0);
}
#endif

// ---------------- prep: concat + fp32->bf16 ----------------
__global__ __launch_bounds__(256) void preph_kernel(const float* __restrict__ hid,
                                                    const float* __restrict__ enc,
                                                    bf16* __restrict__ hb) {
  const size_t i = (size_t)blockIdx.x * 256 + threadIdx.x;
  if (i >= (size_t)S_TOT * DIM) return;
  const int row = (int)(i / DIM);
  const int col = (int)(i % DIM);
  const float v = (row < TXT) ? enc[(size_t)row * DIM + col]
                              : hid[(size_t)(row - TXT) * DIM + col];
  hb[i] = (bf16)v;
}

// ---------------- prep: fold LoRA into weights: Weff = W + up@down ----------------
__global__ __launch_bounds__(256) void prepw_kernel(
    const float* __restrict__ Wq, const float* __restrict__ upq, const float* __restrict__ dnq,
    const float* __restrict__ Wk, const float* __restrict__ upk, const float* __restrict__ dnk,
    const float* __restrict__ Wv, const float* __restrict__ upv, const float* __restrict__ dnv,
    const float* __restrict__ Wo, const float* __restrict__ upo, const float* __restrict__ dno,
    bf16* __restrict__ Wqkv, bf16* __restrict__ Woe) {
  __shared__ float Us[64][65];
  __shared__ float Ds[64][65];
  const int proj = blockIdx.z;
  const float* W; const float* up; const float* dn; bf16* out;
  if (proj == 0)      { W = Wq; up = upq; dn = dnq; out = Wqkv; }
  else if (proj == 1) { W = Wk; up = upk; dn = dnk; out = Wqkv + (size_t)DIM * DIM; }
  else if (proj == 2) { W = Wv; up = upv; dn = dnv; out = Wqkv + (size_t)2 * DIM * DIM; }
  else                { W = Wo; up = upo; dn = dno; out = Woe; }
  const int bm = blockIdx.y * 64;
  const int bn = blockIdx.x * 64;
  const int tx = threadIdx.x & 15;
  const int ty = threadIdx.x >> 4;
  float acc[4][4];
#pragma unroll
  for (int ii = 0; ii < 4; ++ii)
#pragma unroll
    for (int jj = 0; jj < 4; ++jj)
      acc[ii][jj] = W[(size_t)(bm + ty * 4 + ii) * DIM + bn + tx * 4 + jj];
  for (int r0 = 0; r0 < RANK; r0 += 64) {
    for (int i = threadIdx.x; i < 64 * 64; i += 256) {
      const int rr = i >> 6, cc = i & 63;
      Us[rr][cc] = up[(size_t)(bm + rr) * RANK + r0 + cc];
      Ds[rr][cc] = dn[(size_t)(r0 + rr) * DIM + bn + cc];
    }
    __syncthreads();
#pragma unroll 8
    for (int r = 0; r < 64; ++r) {
#pragma unroll
      for (int ii = 0; ii < 4; ++ii)
#pragma unroll
        for (int jj = 0; jj < 4; ++jj)
          acc[ii][jj] += Us[ty * 4 + ii][r] * Ds[r][tx * 4 + jj];
    }
    __syncthreads();
  }
#pragma unroll
  for (int ii = 0; ii < 4; ++ii)
#pragma unroll
    for (int jj = 0; jj < 4; ++jj)
      out[(size_t)(bm + ty * 4 + ii) * DIM + bn + tx * 4 + jj] = (bf16)acc[ii][jj];
}

// ---------------- prep: concat biases ----------------
__global__ __launch_bounds__(256) void prepbias_kernel(const float* __restrict__ bq,
                                                       const float* __restrict__ bk,
                                                       const float* __restrict__ bv,
                                                       float* __restrict__ bqkv) {
  const int i = blockIdx.x * 256 + threadIdx.x;
  if (i >= NQKV) return;
  if (i < DIM)            bqkv[i] = bq[i];
  else if (i < 2 * DIM)   bqkv[i] = bk[i - DIM];
  else                    bqkv[i] = bv[i - 2 * DIM];
}

// ---------------- bf16 WMMA GEMM: out[M,N] = A[M,K] * BT[N,K]^T + bias ----------------
// K = DIM fixed. remap=1 => attention-output row permutation into d_out.
// Double-buffered LDS staging; async DMA global->LDS when available.
__global__ __launch_bounds__(256) void gemm_bf16_kernel(
    const bf16* __restrict__ A, const bf16* __restrict__ BT,
    const float* __restrict__ bias, float* __restrict__ out,
    int ldout, int remap) {
  __shared__ __align__(16) bf16 As[2][128][40];
  __shared__ __align__(16) bf16 Bs[2][128][40];
  const int t     = threadIdx.x;
  const int lane  = t & 31;
  const int wave  = t >> 5;
  const int waveM = wave >> 2;   // 0..1
  const int waveN = wave & 3;    // 0..3
  const int lrow  = lane & 15;
  const int lhalf = lane >> 4;
  const size_t bm = (size_t)blockIdx.y * 128;
  const size_t bn = (size_t)blockIdx.x * 128;

  f32x8 acc[4][2] = {};

  const int srow = t >> 1;
  const int scol = (t & 1) * 16;
  const bf16* ga = A  + (bm + srow) * DIM + scol;
  const bf16* gb = BT + (bn + srow) * DIM + scol;

  auto stage = [&](int k0, int buf) {
#if ASYNC_LDS
    async_ld16(ga + k0,     &As[buf][srow][scol]);
    async_ld16(ga + k0 + 8, &As[buf][srow][scol + 8]);
    async_ld16(gb + k0,     &Bs[buf][srow][scol]);
    async_ld16(gb + k0 + 8, &Bs[buf][srow][scol + 8]);
#else
    bf16x8 a0 = *(const bf16x8*)(ga + k0);
    bf16x8 a1 = *(const bf16x8*)(ga + k0 + 8);
    bf16x8 b0 = *(const bf16x8*)(gb + k0);
    bf16x8 b1 = *(const bf16x8*)(gb + k0 + 8);
    *(bf16x8*)&As[buf][srow][scol]     = a0;
    *(bf16x8*)&As[buf][srow][scol + 8] = a1;
    *(bf16x8*)&Bs[buf][srow][scol]     = b0;
    *(bf16x8*)&Bs[buf][srow][scol + 8] = b1;
#endif
    if (k0 + 64 < DIM) {                       // speculative prefetch 2 stages ahead
      __builtin_prefetch((const void*)(ga + k0 + 64), 0, 0);
      __builtin_prefetch((const void*)(gb + k0 + 64), 0, 0);
    }
  };

  const int NK = DIM / 32;
  stage(0, 0);
  for (int it = 0; it < NK; ++it) {
    if (it + 1 < NK) stage((it + 1) * 32, (it + 1) & 1);
#if ASYNC_LDS
    // ASYNCcnt decrements in issue order: <=4 => this stage's 4 DMAs landed.
    if (it + 1 < NK) asm volatile("s_wait_asynccnt 4" ::: "memory");
    else             asm volatile("s_wait_asynccnt 0" ::: "memory");
#endif
    __syncthreads();
    const int buf = it & 1;
    bf16x16 af[4], bfr[2];
    const int kb = lhalf * 8;
#pragma unroll
    for (int i = 0; i < 4; ++i) {
      const int r = waveM * 64 + i * 16 + lrow;
      af[i] = cat8(*(const bf16x8*)&As[buf][r][kb], *(const bf16x8*)&As[buf][r][kb + 16]);
    }
#pragma unroll
    for (int j = 0; j < 2; ++j) {
      const int r = waveN * 32 + j * 16 + lrow;
      bfr[j] = cat8(*(const bf16x8*)&Bs[buf][r][kb], *(const bf16x8*)&Bs[buf][r][kb + 16]);
    }
#pragma unroll
    for (int i = 0; i < 4; ++i)
#pragma unroll
      for (int j = 0; j < 2; ++j)
        acc[i][j] = wmma_bf16(af[i], bfr[j], acc[i][j]);
    __syncthreads();   // readers of buf done before it is re-staged
  }

#pragma unroll
  for (int i = 0; i < 4; ++i) {
#pragma unroll
    for (int j = 0; j < 2; ++j) {
      const int col = (int)bn + waveN * 32 + j * 16 + lrow;
      const float bv = bias[col];
#pragma unroll
      for (int r = 0; r < 8; ++r) {
        const int row = (int)bm + waveM * 64 + i * 16 + r + lhalf * 8;
        const float v = acc[i][j][r] + bv;
        size_t idx;
        if (remap)
          idx = (row < TXT) ? ((size_t)VID * DIM + (size_t)row * DIM + col)
                            : ((size_t)(row - TXT) * DIM + col);
        else
          idx = (size_t)row * (size_t)ldout + col;
        out[idx] = v;
      }
    }
  }
}

// ---------------- per-head LayerNorm + RoPE, fp32 -> bf16 [head][pos][dim] ----------------
__global__ __launch_bounds__(256) void lnrope_kernel(
    const float* __restrict__ qkv,
    const float* __restrict__ gq, const float* __restrict__ btq,
    const float* __restrict__ gk, const float* __restrict__ btk,
    const float* __restrict__ rc, const float* __restrict__ rs,
    bf16* __restrict__ Qh, bf16* __restrict__ Kh) {
  const int idx = blockIdx.x * 256 + threadIdx.x;
  if (idx >= 2 * NHEAD * S_TOT) return;
  const int pos  = idx & (S_TOT - 1);
  const int head = (idx >> 11) % NHEAD;
  const int sel  = idx / (NHEAD * S_TOT);
  const float* src = qkv + (size_t)pos * NQKV + sel * DIM + head * HDIM;
  float x[HDIM];
  float mu = 0.f;
#pragma unroll
  for (int d = 0; d < HDIM; ++d) { x[d] = src[d]; mu += x[d]; }
  mu *= (1.f / HDIM);
  float var = 0.f;
#pragma unroll
  for (int d = 0; d < HDIM; ++d) { const float c = x[d] - mu; var += c * c; }
  var *= (1.f / HDIM);
  const float inv = rsqrtf(var + 1e-5f);
  const float* g = sel ? gk : gq;
  const float* b = sel ? btk : btq;
#pragma unroll
  for (int d = 0; d < HDIM; ++d) x[d] = (x[d] - mu) * inv * g[d] + b[d];
  if (pos >= TXT) {
    const int vp = pos - TXT;
    const float* c = rc + (size_t)vp * HDIM;
    const float* s = rs + (size_t)vp * HDIM;
#pragma unroll
    for (int j = 0; j < HDIM / 2; ++j) {
      const float x1 = x[2 * j], x2 = x[2 * j + 1];
      x[2 * j]     = x1 * c[2 * j]     - x2 * s[2 * j];
      x[2 * j + 1] = x2 * c[2 * j + 1] + x1 * s[2 * j + 1];
    }
  }
  bf16* dst = (sel ? Kh : Qh) + ((size_t)head * S_TOT + pos) * HDIM;
#pragma unroll
  for (int d = 0; d < HDIM; ++d) dst[d] = (bf16)x[d];
}

// ---------------- V transpose: VT[head*64+dim][pos] bf16 ----------------
__global__ __launch_bounds__(256) void vtrans_kernel(const float* __restrict__ qkv,
                                                     bf16* __restrict__ VT) {
  const size_t i = (size_t)blockIdx.x * 256 + threadIdx.x;
  if (i >= (size_t)DIM * S_TOT) return;
  const int pos = (int)(i & (S_TOT - 1));
  const int d   = (int)(i >> 11);   // 0..3071 = head*64+dim
  VT[(size_t)d * S_TOT + pos] = (bf16)qkv[(size_t)pos * NQKV + 2 * DIM + d];
}

// ---------------- flash attention: one wave = 16 query rows ----------------
__global__ __launch_bounds__(128) void flash_kernel(
    const bf16* __restrict__ Qh, const bf16* __restrict__ Kh,
    const bf16* __restrict__ VT, bf16* __restrict__ attnO) {
  __shared__ __align__(16) bf16 Ps[4][16][72];
  const int head  = blockIdx.x;
  const int lane  = threadIdx.x & 31;
  const int wave  = threadIdx.x >> 5;
  const int lrow  = lane & 15;
  const int lhalf = lane >> 4;
  const int qbase = blockIdx.y * 64 + wave * 16;

  const bf16* Qb = Qh + ((size_t)head * S_TOT + qbase) * HDIM;
  const bf16* Kb = Kh + (size_t)head * S_TOT * HDIM;
  const bf16* Vb = VT + (size_t)head * HDIM * S_TOT;

  bf16x16 qa[2];
#pragma unroll
  for (int ks = 0; ks < 2; ++ks) {
    const bf16* p = Qb + lrow * HDIM + ks * 32 + lhalf * 8;
    qa[ks] = cat8(*(const bf16x8*)p, *(const bf16x8*)(p + 16));
  }

  f32x8 O[4] = {};
  float ms[8], ls[8];
#pragma unroll
  for (int r = 0; r < 8; ++r) { ms[r] = -1e30f; ls[r] = 0.f; }

  for (int kc = 0; kc < S_TOT; kc += 64) {
    f32x8 st[4];
#pragma unroll
    for (int tl = 0; tl < 4; ++tl) {
      f32x8 s = {};
#pragma unroll
      for (int ks = 0; ks < 2; ++ks) {
        const bf16* kp = Kb + (size_t)(kc + tl * 16 + lrow) * HDIM + ks * 32 + lhalf * 8;
        bf16x16 kf = cat8(*(const bf16x8*)kp, *(const bf16x8*)(kp + 16));
        s = wmma_bf16(qa[ks], kf, s);
      }
#pragma unroll
      for (int r = 0; r < 8; ++r) s[r] *= 0.125f;   // 1/sqrt(64)
      st[tl] = s;
    }
    float corr[8];
#pragma unroll
    for (int r = 0; r < 8; ++r) {
      float mx = st[0][r];
      mx = fmaxf(mx, st[1][r]); mx = fmaxf(mx, st[2][r]); mx = fmaxf(mx, st[3][r]);
      mx = redmax16(mx);
      const float mnew = fmaxf(ms[r], mx);
      corr[r] = __expf(ms[r] - mnew);
      ms[r] = mnew;
    }
    float rsum[8];
#pragma unroll
    for (int r = 0; r < 8; ++r) rsum[r] = 0.f;
#pragma unroll
    for (int tl = 0; tl < 4; ++tl) {
#pragma unroll
      for (int r = 0; r < 8; ++r) {
        const float p = __expf(st[tl][r] - ms[r]);
        rsum[r] += p;
        Ps[wave][r + lhalf * 8][tl * 16 + lrow] = (bf16)p;  // C-layout -> row-major tile
      }
    }
#pragma unroll
    for (int r = 0; r < 8; ++r) {
      ls[r] = ls[r] * corr[r] + redsum16(rsum[r]);
#pragma unroll
      for (int tl = 0; tl < 4; ++tl) O[tl][r] *= corr[r];
    }
    asm volatile("s_wait_dscnt 0" ::: "memory");  // CDNA5 split DS counter: LDS RAW in-wave
    bf16x16 pa[2];
#pragma unroll
    for (int ks = 0; ks < 2; ++ks) {
      const int c = ks * 32 + lhalf * 8;
      pa[ks] = cat8(*(const bf16x8*)&Ps[wave][lrow][c], *(const bf16x8*)&Ps[wave][lrow][c + 16]);
    }
#pragma unroll
    for (int tl = 0; tl < 4; ++tl) {
#pragma unroll
      for (int ks = 0; ks < 2; ++ks) {
        const bf16* vp = Vb + (size_t)(tl * 16 + lrow) * S_TOT + kc + ks * 32 + lhalf * 8;
        bf16x16 vf = cat8(*(const bf16x8*)vp, *(const bf16x8*)(vp + 16));
        O[tl] = wmma_bf16(pa[ks], vf, O[tl]);
      }
    }
  }
#pragma unroll
  for (int tl = 0; tl < 4; ++tl) {
#pragma unroll
    for (int r = 0; r < 8; ++r) {
      const int row = qbase + r + lhalf * 8;
      const int col = head * HDIM + tl * 16 + lrow;
      attnO[(size_t)row * DIM + col] = (bf16)(O[tl][r] / ls[r]);
    }
  }
}

// ---------------- launch ----------------
extern "C" void kernel_launch(void* const* d_in, const int* in_sizes, int n_in,
                              void* d_out, int out_size, void* d_ws, size_t ws_size,
                              hipStream_t stream) {
  (void)in_sizes; (void)n_in; (void)out_size; (void)ws_size;
  const float* hid  = (const float*)d_in[0];
  const float* enc  = (const float*)d_in[1];
  const float* rc   = (const float*)d_in[2];
  const float* rs   = (const float*)d_in[3];
  const float* Wq   = (const float*)d_in[4];
  const float* bq   = (const float*)d_in[5];
  const float* Wk   = (const float*)d_in[6];
  const float* bk   = (const float*)d_in[7];
  const float* Wv   = (const float*)d_in[8];
  const float* bv   = (const float*)d_in[9];
  const float* Wo   = (const float*)d_in[10];
  const float* bo   = (const float*)d_in[11];
  const float* lqd  = (const float*)d_in[12];
  const float* lqu  = (const float*)d_in[13];
  const float* lkd  = (const float*)d_in[14];
  const float* lku  = (const float*)d_in[15];
  const float* lvd  = (const float*)d_in[16];
  const float* lvu  = (const float*)d_in[17];
  const float* lpd  = (const float*)d_in[18];
  const float* lpu  = (const float*)d_in[19];
  const float* gq   = (const float*)d_in[20];
  const float* btq  = (const float*)d_in[21];
  const float* gk   = (const float*)d_in[22];
  const float* btk  = (const float*)d_in[23];

  char* ws = (char*)d_ws;
  size_t o = 0;
  auto carve = [&](size_t bytes) -> void* {
    o = (o + 255) & ~(size_t)255;
    void* p = ws + o;
    o += bytes;
    return p;
  };
  bf16*  hb     = (bf16*) carve((size_t)S_TOT * DIM * sizeof(bf16));
  bf16*  Wqkv   = (bf16*) carve((size_t)3 * DIM * DIM * sizeof(bf16));
  bf16*  Woe    = (bf16*) carve((size_t)DIM * DIM * sizeof(bf16));
  float* bqkv   = (float*)carve((size_t)NQKV * sizeof(float));
  float* qkvbuf = (float*)carve((size_t)S_TOT * NQKV * sizeof(float));
  bf16*  Qh     = (bf16*) carve((size_t)NHEAD * S_TOT * HDIM * sizeof(bf16));
  bf16*  Kh     = (bf16*) carve((size_t)NHEAD * S_TOT * HDIM * sizeof(bf16));
  bf16*  VT     = (bf16*) carve((size_t)NHEAD * HDIM * S_TOT * sizeof(bf16));
  bf16*  attn   = (bf16*) carve((size_t)S_TOT * DIM * sizeof(bf16));

  preph_kernel<<<(S_TOT * DIM + 255) / 256, 256, 0, stream>>>(hid, enc, hb);
  prepw_kernel<<<dim3(DIM / 64, DIM / 64, 4), 256, 0, stream>>>(
      Wq, lqu, lqd, Wk, lku, lkd, Wv, lvu, lvd, Wo, lpu, lpd, Wqkv, Woe);
  prepbias_kernel<<<(NQKV + 255) / 256, 256, 0, stream>>>(bq, bk, bv, bqkv);
  gemm_bf16_kernel<<<dim3(NQKV / 128, S_TOT / 128), 256, 0, stream>>>(
      hb, Wqkv, bqkv, qkvbuf, NQKV, 0);
  lnrope_kernel<<<(2 * NHEAD * S_TOT + 255) / 256, 256, 0, stream>>>(
      qkvbuf, gq, btq, gk, btk, rc, rs, Qh, Kh);
  vtrans_kernel<<<(int)(((size_t)DIM * S_TOT + 255) / 256), 256, 0, stream>>>(qkvbuf, VT);
  flash_kernel<<<dim3(NHEAD, S_TOT / 64), 128, 0, stream>>>(Qh, Kh, VT, attn);
  gemm_bf16_kernel<<<dim3(DIM / 128, S_TOT / 128), 256, 0, stream>>>(
      attn, Woe, bo, (float*)d_out, DIM, 1);
}
